// LSTM_81819126989241
// MI455X (gfx1250) — matrix operand order repaired and checked by reading, not compiled, not run on hardware
//
#include <hip/hip_runtime.h>

// ---------------------------------------------------------------------------
// LSTM (B=64, T=2048, D=F=128) for gfx1250 (MI455X).
//
// Phase 1 (parallel, memory-bound): Z[b,t,:] = x[b,t,:] @ Wi + bias, computed
//   with v_wmma_f32_16x16x32_bf16 and stored in *accumulator-register layout*
//   (per (g,t,ntile) tile: 32 lanes x 8 floats contiguous) so phase 2 can
//   reload it as the WMMA C operand with two b128 loads per tile.
// Phase 2 (sequential scan, latency-bound): 4 workgroups (one 16-row batch
//   tile each, one WGP each). Wh^T lives in LDS as bf16 (139 KB, CDNA5-only:
//   320 KB/WGP). c-state lives in VGPRs for all 2048 steps. h is double-
//   buffered in LDS (bf16) -> exactly one s_barrier per timestep.
// ---------------------------------------------------------------------------

#define DEVINL __device__ __forceinline__

typedef __attribute__((ext_vector_type(16))) __bf16  v16bf;
typedef __attribute__((ext_vector_type(8)))  float   v8f;
typedef __attribute__((ext_vector_type(4)))  float   f4;
typedef __attribute__((ext_vector_type(4)))  unsigned int u4;

struct U4x2 { u4 a, b; };
struct F4x2 { f4 a, b; };

static constexpr int BB = 64, TT = 2048, DD = 128, FF = 128;
static constexpr int GD = 4 * FF;        // fused gate dim = 512
static constexpr int NT = GD / 16;       // 32 n-tiles of width 16
static constexpr int WS_ = 136;          // padded LDS row stride (bf16 elems)
static constexpr int HS_ = 136;          // padded h row stride
static constexpr int HBUF = 16 * HS_;    // one h buffer (ushorts)

DEVINL unsigned short f2bf_u(float f) {          // f32 -> bf16 bits, RNE
  unsigned u = __builtin_bit_cast(unsigned, f);
  u += 0x7FFFu + ((u >> 16) & 1u);
  return (unsigned short)(u >> 16);
}
DEVINL __bf16 f2bf(float f) {
  unsigned short s = f2bf_u(f);
  return __builtin_bit_cast(__bf16, s);
}
DEVINL v16bf ld2x8(const unsigned short* p0, const unsigned short* p1) {
  U4x2 u; u.a = *(const u4*)p0; u.b = *(const u4*)p1;   // 2x ds/global b128
  return __builtin_bit_cast(v16bf, u);
}
DEVINL v8f ld_v8f(const float* p) {
  F4x2 r; r.a = *(const f4*)p; r.b = *(const f4*)(p + 4);
  return __builtin_bit_cast(v8f, r);
}
DEVINL float sigm_f(float v) {                 // 1/(1+e^-x) via v_exp + v_rcp
  return __builtin_amdgcn_rcpf(1.0f + __expf(-v));
}
DEVINL float tanh_f(float v) {                 // 1 - 2/(e^{2x}+1)
  return 1.0f - 2.0f * __builtin_amdgcn_rcpf(__expf(2.0f * v) + 1.0f);
}

// ===========================================================================
// Phase 1: Z = x @ Wi + bias  (grid: (T/8, B/16); wave w handles t = bx*8+w)
// ===========================================================================
__global__ void __launch_bounds__(256)
lstm_phase1(const float* __restrict__ x, const float* __restrict__ Wi,
            const float* __restrict__ bias, float* __restrict__ Z)
{
  extern __shared__ unsigned short lds[];
  unsigned short* WiT = lds;                       // [GD][WS_] Wi^T, bf16

  const int tid = threadIdx.x;
  for (int idx = tid; idx < DD * GD; idx += 256) { // coalesced global reads
    int k = idx / GD, n = idx % GD;                // Wi[k][n]
    WiT[n * WS_ + k] = f2bf_u(Wi[idx]);
  }
  __syncthreads();

  const int w = tid >> 5, l = tid & 31;
  const int t = blockIdx.x * 8 + w;
  const int g = blockIdx.y;                        // batch tile (16 rows)
  const int m  = l & 15;                           // A-matrix row
  const int k0 = (l < 16) ? 0 : 8;                 // A-matrix K sub-run

  // Build A (x row for (16g+m, t)) as 4 blocks of 16x32 bf16
  const float* xr = x + ((size_t)(16 * g + m) * TT + t) * DD;
  v16bf a[4];
#pragma unroll
  for (int kb = 0; kb < 4; ++kb) {
    const float* p1 = xr + 32 * kb + k0;           // K run1: 8 consecutive f32
    const float* p2 = p1 + 16;                     // K run2
    f4 r1a = *(const f4*)p1, r1b = *(const f4*)(p1 + 4);
    f4 r2a = *(const f4*)p2, r2b = *(const f4*)(p2 + 4);
    v16bf av;
#pragma unroll
    for (int i = 0; i < 4; ++i) {
      av[i]      = f2bf(r1a[i]); av[4 + i]  = f2bf(r1b[i]);
      av[8 + i]  = f2bf(r2a[i]); av[12 + i] = f2bf(r2b[i]);
    }
    a[kb] = av;
  }

  const size_t zb = (((size_t)g * TT + t) * NT) * 256 + (size_t)l * 8;
#pragma unroll 1
  for (int nt = 0; nt < NT; ++nt) {
    const int n0 = nt * 16;
    const float bv = bias[n0 + (l & 15)];
    v8f acc;
#pragma unroll
    for (int v = 0; v < 8; ++v) acc[v] = bv;
#pragma unroll
    for (int kb = 0; kb < 4; ++kb) {
      const unsigned short* bp =
          WiT + (n0 + (l & 15)) * WS_ + 32 * kb + ((l >> 4) << 3);
      acc = __builtin_amdgcn_wmma_f32_16x16x32_bf16(
          false, a[kb], false, ld2x8(bp, bp + 16), (short)0, acc, false, false);
    }
    F4x2 st = __builtin_bit_cast(F4x2, acc);       // store in C-register layout
    float* zp = Z + zb + (size_t)nt * 256;
    *(f4*)zp = st.a; *(f4*)(zp + 4) = st.b;
  }
}

// ===========================================================================
// Phase 2: sequential scan. grid = B/16 blocks, 256 threads (8 waves).
// Wave w owns gate columns 16w..16w+15 of all 4 gates + c-state in VGPRs.
// ===========================================================================
template <bool USE_Z>
__global__ void __launch_bounds__(256)
lstm_phase2(const float* __restrict__ x, const float* __restrict__ h0,
            const float* __restrict__ c0, const float* __restrict__ Wi,
            const float* __restrict__ Wh, const float* __restrict__ bias,
            const float* __restrict__ Z, float* __restrict__ out)
{
  extern __shared__ unsigned short lds[];
  unsigned short* WhT = lds;                                   // [GD][WS_]
  unsigned short* WiT = USE_Z ? nullptr : (lds + GD * WS_);    // fused fallback
  unsigned short* hb  = lds + (USE_Z ? GD * WS_ : 2 * GD * WS_); // 2x[16][HS_]

  const int tid = threadIdx.x;
  const int g = blockIdx.x;

  for (int idx = tid; idx < FF * GD; idx += 256) {
    int k = idx / GD, n = idx % GD;
    WhT[n * WS_ + k] = f2bf_u(Wh[idx]);
    if constexpr (!USE_Z) WiT[n * WS_ + k] = f2bf_u(Wi[idx]);
  }
  for (int idx = tid; idx < 16 * FF; idx += 256) {             // h0 -> buf 0
    int mm = idx / FF, nc = idx % FF;
    hb[mm * HS_ + nc] = f2bf_u(h0[(size_t)(16 * g + mm) * FF + nc]);
  }
  __syncthreads();

  const int w = tid >> 5, l = tid & 31;
  const int hi = l >> 4;
  const int nn = l & 15;                    // C-layout col / B-layout col
  const int m  = l & 15;                    // A-layout row
  const int k0 = hi ? 8 : 0;
  const int col = 16 * w + nn;              // owned feature column

  v8f creg;                                 // persistent cell state (f32)
#pragma unroll
  for (int v = 0; v < 8; ++v)
    creg[v] = c0[(size_t)(16 * g + v + 8 * hi) * FF + col];

  const size_t orow = (size_t)TT * FF;
  float* ob = out + (size_t)(16 * g + 8 * hi) * orow + col;

  float bgate[4];
#pragma unroll
  for (int G = 0; G < 4; ++G) bgate[G] = bias[G * FF + col];

#pragma unroll 1
  for (int t = 0; t < TT; ++t) {
    const int cur = t & 1, nxt = cur ^ 1;
    const unsigned short* hc = hb + cur * HBUF;

    // ---- accumulator init: Z slice (or bias splat in fused mode) ----
    v8f zacc[4];
    if constexpr (USE_Z) {
      const float* zt = Z + ((size_t)g * TT + t) * ((size_t)NT * 256)
                          + (size_t)l * 8;
#pragma unroll
      for (int G = 0; G < 4; ++G)
        zacc[G] = ld_v8f(zt + (size_t)(G * 8 + w) * 256);
      if (t + 1 < TT) {                      // global_prefetch next-t Z slice
        const float* zn = zt + (size_t)NT * 256;
#pragma unroll
        for (int G = 0; G < 4; ++G)
          __builtin_prefetch(zn + (size_t)(G * 8 + w) * 256, 0, 1);
      }
    } else {
#pragma unroll
      for (int G = 0; G < 4; ++G) {
        v8f s;
#pragma unroll
        for (int v = 0; v < 8; ++v) s[v] = bgate[G];
        zacc[G] = s;
      }
    }

    // ---- A = h_t (16x128 bf16) from LDS, 8x ds_load_b128 ----
    v16bf a[4];
#pragma unroll
    for (int kb = 0; kb < 4; ++kb) {
      const unsigned short* p = hc + m * HS_ + 32 * kb + k0;
      a[kb] = ld2x8(p, p + 16);
    }
    v16bf xa[4];
    if constexpr (!USE_Z) {                  // fused: also A = x_t from global
      const float* xr = x + ((size_t)(16 * g + m) * TT + t) * DD;
#pragma unroll
      for (int kb = 0; kb < 4; ++kb) {
        const float* p1 = xr + 32 * kb + k0;
        const float* p2 = p1 + 16;
        f4 r1a = *(const f4*)p1, r1b = *(const f4*)(p1 + 4);
        f4 r2a = *(const f4*)p2, r2b = *(const f4*)(p2 + 4);
        v16bf av;
#pragma unroll
        for (int i = 0; i < 4; ++i) {
          av[i]      = f2bf(r1a[i]); av[4 + i]  = f2bf(r1b[i]);
          av[8 + i]  = f2bf(r2a[i]); av[12 + i] = f2bf(r2b[i]);
        }
        xa[kb] = av;
      }
    }

    // ---- 4 gates x (K=128) WMMA: h@Wh (+ x@Wi in fused mode) ----
#pragma unroll
    for (int G = 0; G < 4; ++G) {
      const int n0 = (G * 8 + w) * 16;
      v8f acc = zacc[G];
#pragma unroll
      for (int kb = 0; kb < 4; ++kb) {
        const unsigned short* bp = WhT + (n0 + nn) * WS_ + 32 * kb + 8 * hi;
        acc = __builtin_amdgcn_wmma_f32_16x16x32_bf16(
            false, a[kb], false, ld2x8(bp, bp + 16), (short)0, acc, false, false);
      }
      if constexpr (!USE_Z) {
#pragma unroll
        for (int kb = 0; kb < 4; ++kb) {
          const unsigned short* bp = WiT + (n0 + nn) * WS_ + 32 * kb + 8 * hi;
          acc = __builtin_amdgcn_wmma_f32_16x16x32_bf16(
              false, xa[kb], false, ld2x8(bp, bp + 16), (short)0, acc, false, false);
        }
      }
      zacc[G] = acc;
    }

    // ---- gate nonlinearity, c/h update, write out + next h buffer ----
    unsigned short* hn = hb + nxt * HBUF;
#pragma unroll
    for (int v = 0; v < 8; ++v) {
      float iv = sigm_f(zacc[0][v]);
      float fv = sigm_f(zacc[1][v]);
      float gv = tanh_f(zacc[2][v]);
      float ov = sigm_f(zacc[3][v]);
      float cv = fv * creg[v] + iv * gv;
      creg[v] = cv;
      float hv = ov * tanh_f(cv);
      ob[(size_t)v * orow + (size_t)t * FF] = hv;      // out[b, t, col]
      hn[(v + 8 * hi) * HS_ + col] = f2bf_u(hv);       // h_{t+1} (bf16, LDS)
    }
    __syncthreads();   // one barrier per step (double-buffered h)
  }
}

// ===========================================================================
extern "C" void kernel_launch(void* const* d_in, const int* in_sizes, int n_in,
                              void* d_out, int out_size, void* d_ws, size_t ws_size,
                              hipStream_t stream) {
  (void)in_sizes; (void)n_in; (void)out_size;
  const float* x    = (const float*)d_in[0];
  const float* h0   = (const float*)d_in[1];
  const float* c0   = (const float*)d_in[2];
  const float* Wi   = (const float*)d_in[3];
  const float* Wh   = (const float*)d_in[4];
  const float* bias = (const float*)d_in[5];
  float* out = (float*)d_out;
  float* Z   = (float*)d_ws;

  const size_t zbytes = (size_t)BB * TT * GD * sizeof(float);   // 256 MB
  const size_t ldsW   = (size_t)GD * WS_ * sizeof(unsigned short); // 139264 B
  const size_t ldsH   = (size_t)2 * HBUF * sizeof(unsigned short);

  if (ws_size >= zbytes) {
    // Phase 1: full-chip parallel GEMM (memory-bound, ~15 us at 23.3 TB/s)
    lstm_phase1<<<dim3(TT / 8, BB / 16), 256, ldsW, stream>>>(x, Wi, bias, Z);
    // Phase 2: 4 WGPs run the batch-split recurrence
    lstm_phase2<true><<<dim3(BB / 16), 256, ldsW + ldsH, stream>>>(
        x, h0, c0, Wi, Wh, bias, Z, out);
  } else {
    // Fallback: fused x@Wi inside the scan (Wi^T also LDS-resident; 287 KB)
    lstm_phase2<false><<<dim3(BB / 16), 256, 2 * ldsW + ldsH, stream>>>(
        x, h0, c0, Wi, Wh, bias, Z, out);
  }
}
